// ParallelExpertAttention_8564164789006
// MI455X (gfx1250) — matrix operand order repaired and checked by reading, not compiled, not run on hardware
//
#include <hip/hip_runtime.h>
#include <hip/hip_bf16.h>

typedef _Float16 f16;
typedef __attribute__((ext_vector_type(16))) _Float16 v16h;
typedef __attribute__((ext_vector_type(8)))  _Float16 v8h;
typedef __attribute__((ext_vector_type(8)))  float    v8f;

#define Bc 2
#define Sc 2048
#define Hc 768
#define Ec 4
#define NHc 12
#define WINc 256
#define HDc 64
#define Mc (Bc * Sc)          // 4096
#define EHc (Ec * Hc)         // 3072

// ---------- fragment loaders (contiguous k = 16*(lane/16) + s, same order for A and B) ----------
// A matrix 16x32: lanes 0-15 hold rows M=0..15 with K = {0..15};
// lanes 16-31 hold the same rows with K = {16..31}.
__device__ __forceinline__ v16h load_a_frag(const f16* rowbase, int lane) {
    union { v16h v; v8h h[2]; } u;
    const int o = ((lane >> 4) << 4);
    u.h[0] = *(const v8h*)(rowbase + o);
    u.h[1] = *(const v8h*)(rowbase + 8 + o);
    return u.v;
}

#define WMMA_F16(A, Bf, C) \
    __builtin_amdgcn_wmma_f32_16x16x32_f16(false, (A), false, (Bf), (short)0, (C), false, false)

// ---------- async copy helpers (Tensor/async data path, tracked by ASYNCcnt) ----------
__device__ __forceinline__ void async_ld_b128(unsigned ldsa, const void* ga) {
    asm volatile("global_load_async_to_lds_b128 %0, %1, off" :: "v"(ldsa), "v"(ga) : "memory");
}
__device__ __forceinline__ void async_ld_b128_o32(unsigned ldsa, const void* ga) {
    asm volatile("global_load_async_to_lds_b128 %0, %1, off offset:32" :: "v"(ldsa), "v"(ga) : "memory");
}
#define WAIT_ASYNC_2() asm volatile("s_wait_asynccnt 0x2" ::: "memory")
#define WAIT_ASYNC_0() asm volatile("s_wait_asynccnt 0x0" ::: "memory")
#define WAIT_DS_0()    asm volatile("s_wait_dscnt 0x0" ::: "memory")

// ---------- f32 -> f16 conversion ----------
__global__ void cvt_f32_f16_kernel(const float* __restrict__ s, f16* __restrict__ d, long long n) {
    long long i = (long long)blockIdx.x * blockDim.x + threadIdx.x;
    long long stride = (long long)gridDim.x * blockDim.x;
    for (; i < n; i += stride) d[i] = (f16)s[i];
}

// ---------- V transpose: v[e][b*S+s][h*64+d] -> vt[((e*B+b)*NH+h)*HD+d][s] ----------
__global__ void transpose_v_kernel(const f16* __restrict__ V, f16* __restrict__ VT) {
    const long long total = (long long)Ec * Bc * NHc * HDc * Sc;
    long long i = (long long)blockIdx.x * blockDim.x + threadIdx.x;
    long long stride = (long long)gridDim.x * blockDim.x;
    for (; i < total; i += stride) {
        int s = (int)(i & (Sc - 1));
        long long r = i >> 11;
        int d = (int)(r & (HDc - 1)); r >>= 6;
        int h = (int)(r % NHc); r /= NHc;
        int b = (int)(r & 1);
        int e = (int)(r >> 1);
        VT[i] = V[(size_t)e * ((size_t)Mc * Hc) + (size_t)(b * Sc + s) * Hc + h * HDc + d];
    }
}

// ---------- WMMA GEMM with async-to-LDS staged B tiles ----------
// C[m][n] = sum_k A[m][k] * B[n][k]; f16 in/out, f32 acc.
// block = 256 threads = 8 waves; wave tile 16x16; block tile 32 rows x 64 cols.
// Each wave double-buffers its private 16x32 B tile in LDS via
// GLOBAL_LOAD_ASYNC_TO_LDS_B128 (ASYNCcnt pipeline), A frags load direct.
__global__ void gemm_f16_kernel(const f16* __restrict__ A, const f16* __restrict__ B,
                                f16* __restrict__ C, int K, int lda, int ldb, int ldc,
                                long long sA, long long sB, long long sC) {
    __shared__ __align__(128) f16 bstage[8][2][16 * 32];   // 16 KB: 8 waves x 2 bufs x 1KB
    const int lane = threadIdx.x & 31;
    const int wid  = threadIdx.x >> 5;
    const int z    = blockIdx.z;
    const int m0   = blockIdx.y * 32 + (wid >> 2) * 16;
    const int n0   = blockIdx.x * 64 + (wid & 3) * 16;

    const f16* arow = A + (size_t)z * sA + (size_t)(m0 + (lane & 15)) * lda;
    // per-lane global source for async staging: row = n0 + lane&15, 8-half chunk = lane>>4
    const f16* gb   = B + (size_t)z * sB + (size_t)(n0 + (lane & 15)) * ldb + ((lane >> 4) << 3);
    // per-lane LDS dest: row*64B + chunk*16B (both LDS and global get IOFFSET, so one addr serves 2 ops)
    const unsigned ldsbase = (unsigned)(uintptr_t)(&bstage[wid][0][0]);
    const unsigned laneoff = (unsigned)((lane & 15) * 64 + ((lane >> 4) << 4));
    const f16* lb = &bstage[wid][0][0] + (size_t)(lane & 15) * 32 + ((lane >> 4) << 4);

    v8f acc = {};
    // prologue: stage first B tile (2 async b128 per wave = 1KB tile)
    async_ld_b128(ldsbase + laneoff, gb);
    async_ld_b128_o32(ldsbase + laneoff, gb);

    int kk = 0, buf = 0;
    for (; kk + 32 < K; kk += 32, buf ^= 1) {
        // make sure previous consumers of the buffer we are about to overwrite retired
        WAIT_DS_0();
        const unsigned nl = ldsbase + (unsigned)((buf ^ 1) * 1024) + laneoff;
        async_ld_b128(nl, gb + kk + 32);
        async_ld_b128_o32(nl, gb + kk + 32);
        __builtin_prefetch(arow + kk + 256, 0, 1);
        WAIT_ASYNC_2();                       // current buffer's 2 ops done (in-order)
        v16h a  = load_a_frag(arow + kk, lane);
        v16h bb = *(const v16h*)(lb + (size_t)buf * 512);
        acc = WMMA_F16(a, bb, acc);
    }
    WAIT_ASYNC_0();
    v16h a  = load_a_frag(arow + kk, lane);
    v16h bb = *(const v16h*)(lb + (size_t)buf * 512);
    acc = WMMA_F16(a, bb, acc);

    f16* cp = C + (size_t)z * sC + (size_t)(m0 + ((lane >> 4) << 3)) * ldc + n0 + (lane & 15);
#pragma unroll
    for (int i = 0; i < 8; i++) cp[(size_t)i * ldc] = (f16)acc[i];
}

// ---------- sliding-window flash attention ----------
// one wave per (e, b, h, 16-query tile); 32-key blocks.
__global__ void attn_kernel(const f16* __restrict__ Q, const f16* __restrict__ K,
                            const f16* __restrict__ VT, f16* __restrict__ O) {
    __shared__ __align__(64) f16 pbuf[4][16 * 32];
    const int lane = threadIdx.x & 31;
    const int wid  = threadIdx.x >> 5;
    int t = blockIdx.x * 4 + wid;
    const int qt = t & 127; t >>= 7;
    const int h  = t % NHc; t /= NHc;
    const int b  = t & 1;
    const int e  = t >> 1;
    const int q0 = qt << 4;

    const size_t mh = (size_t)Mc * Hc;
    const f16* Qe  = Q + (size_t)e * mh;
    const f16* Ke  = K + (size_t)e * mh;
    const f16* Vth = VT + ((size_t)((e * Bc + b) * NHc + h) * HDc) * Sc;

    const f16* qrow = Qe + (size_t)(b * Sc + q0 + (lane & 15)) * Hc + h * HDc;
    v16h aq0 = load_a_frag(qrow, lane);
    v16h aq1 = load_a_frag(qrow + 32, lane);

    v8f o0 = {}, o1 = {}, o2 = {}, o3 = {};
    float mrow[8], lrow[8];
#pragma unroll
    for (int i = 0; i < 8; i++) { mrow[i] = -1e30f; lrow[i] = 0.0f; }
    const float scale = 0.125f;  // 1/sqrt(64)

    int kstart = q0 - (WINc - 1); if (kstart < 0) kstart = 0;
    kstart &= ~31;
    const int kend = q0 + 15;
    f16* pb = pbuf[wid];

    for (int kb = kstart; kb <= kend; kb += 32) {
        // ---- S = Q K^T (two 16x16 column tiles, contraction d=64 in two x32 steps) ----
        const f16* k0p = Ke + (size_t)(b * Sc + kb + (lane & 15)) * Hc + h * HDc + ((lane >> 4) << 4);
        const f16* k1p = k0p + (size_t)16 * Hc;
        v16h bk00 = *(const v16h*)(k0p);
        v16h bk01 = *(const v16h*)(k0p + 32);
        v16h bk10 = *(const v16h*)(k1p);
        v16h bk11 = *(const v16h*)(k1p + 32);
        v8f s0 = {}, s1 = {};
        s0 = WMMA_F16(aq0, bk00, s0);
        s0 = WMMA_F16(aq1, bk01, s0);
        s1 = WMMA_F16(aq0, bk10, s1);
        s1 = WMMA_F16(aq1, bk11, s1);

        // ---- masked online softmax (row stats replicated across 16-lane halves) ----
        const int col0 = kb + (lane & 15);
        const int col1 = col0 + 16;
        float p0[8], p1[8];
#pragma unroll
        for (int i = 0; i < 8; i++) {
            const int qi = q0 + i + ((lane >> 4) << 3);
            const int d0 = qi - col0, d1 = qi - col1;
            const bool ok0 = (d0 >= 0) && (d0 < WINc);
            const bool ok1 = (d1 >= 0) && (d1 < WINc);
            float x0 = ok0 ? s0[i] * scale : -1e30f;
            float x1 = ok1 ? s1[i] * scale : -1e30f;
            float mx = fmaxf(x0, x1);
#pragma unroll
            for (int off = 1; off < 16; off <<= 1) mx = fmaxf(mx, __shfl_xor(mx, off, 32));
            const float mn   = fmaxf(mrow[i], mx);
            const float corr = __expf(mrow[i] - mn);
            const float e0 = ok0 ? __expf(x0 - mn) : 0.0f;
            const float e1 = ok1 ? __expf(x1 - mn) : 0.0f;
            float rs = e0 + e1;
#pragma unroll
            for (int off = 1; off < 16; off <<= 1) rs += __shfl_xor(rs, off, 32);
            lrow[i] = lrow[i] * corr + rs;
            mrow[i] = mn;
            o0[i] *= corr; o1[i] *= corr; o2[i] *= corr; o3[i] *= corr;
            p0[i] = e0; p1[i] = e1;
        }

        // ---- re-layout P (C-layout -> A-fragment) through LDS ----
#pragma unroll
        for (int i = 0; i < 8; i++) {
            const int r = i + ((lane >> 4) << 3);
            pb[r * 32 + (lane & 15)]      = (f16)p0[i];
            pb[r * 32 + 16 + (lane & 15)] = (f16)p1[i];
        }
        WAIT_DS_0();
        v16h ap = load_a_frag(pb + (size_t)(lane & 15) * 32, lane);

        // ---- O += P V (V transposed: contiguous over keys) ----
        const f16* vbase = Vth + (size_t)(lane & 15) * Sc + kb + ((lane >> 4) << 4);
        v16h bv0 = *(const v16h*)(vbase);
        v16h bv1 = *(const v16h*)(vbase + (size_t)16 * Sc);
        v16h bv2 = *(const v16h*)(vbase + (size_t)32 * Sc);
        v16h bv3 = *(const v16h*)(vbase + (size_t)48 * Sc);
        o0 = WMMA_F16(ap, bv0, o0);
        o1 = WMMA_F16(ap, bv1, o1);
        o2 = WMMA_F16(ap, bv2, o2);
        o3 = WMMA_F16(ap, bv3, o3);
    }

    // ---- normalize + store ----
#pragma unroll
    for (int i = 0; i < 8; i++) {
        const float inv = 1.0f / lrow[i];
        const int m = b * Sc + q0 + i + ((lane >> 4) << 3);
        f16* op = O + (size_t)e * mh + (size_t)m * Hc + h * HDc + (lane & 15);
        op[0]  = (f16)(o0[i] * inv);
        op[16] = (f16)(o1[i] * inv);
        op[32] = (f16)(o2[i] * inv);
        op[48] = (f16)(o3[i] * inv);
    }
}

// ---------- fused gated MLP: out = silu(F Wg^T) * (F Wu^T), f32 out ----------
__global__ void mlp_kernel(const f16* __restrict__ F, const f16* __restrict__ WG,
                           const f16* __restrict__ WU, float* __restrict__ OUT) {
    const int lane = threadIdx.x & 31;
    const int wid  = threadIdx.x >> 5;
    const int m0   = blockIdx.y * 32 + (wid >> 2) * 16;
    const int n0   = blockIdx.x * 64 + (wid & 3) * 16;
    const f16* arow = F  + (size_t)(m0 + (lane & 15)) * EHc;
    const f16* gcol = WG + (size_t)(n0 + (lane & 15)) * EHc + ((lane >> 4) << 4);
    const f16* ucol = WU + (size_t)(n0 + (lane & 15)) * EHc + ((lane >> 4) << 4);
    v8f ag = {}, au = {};
#pragma unroll 4
    for (int kk = 0; kk < EHc; kk += 32) {
        __builtin_prefetch(arow + kk + 256, 0, 1);
        v16h a = load_a_frag(arow + kk, lane);
        ag = WMMA_F16(a, *(const v16h*)(gcol + kk), ag);
        au = WMMA_F16(a, *(const v16h*)(ucol + kk), au);
    }
    float* op = OUT + (size_t)(m0 + ((lane >> 4) << 3)) * Hc + n0 + (lane & 15);
#pragma unroll
    for (int i = 0; i < 8; i++) {
        const float g = ag[i];
        const float u = au[i];
        const float sg = g / (1.0f + __expf(-g));
        op[(size_t)i * Hc] = sg * u;
    }
}

// ---------- host-side orchestration ----------
extern "C" void kernel_launch(void* const* d_in, const int* in_sizes, int n_in,
                              void* d_out, int out_size, void* d_ws, size_t ws_size,
                              hipStream_t stream) {
    const float* x      = (const float*)d_in[0];
    const float* wq     = (const float*)d_in[1];
    const float* wk     = (const float*)d_in[2];
    const float* wv     = (const float*)d_in[3];
    const float* wo     = (const float*)d_in[4];
    const float* w_gate = (const float*)d_in[5];
    const float* w_up   = (const float*)d_in[6];
    float* out = (float*)d_out;

    // f16 workspace layout
    const long long XB  = (long long)Mc * Hc;           // 3,145,728
    const long long WEB = (long long)Ec * Hc * Hc;      // 2,359,296
    const long long WMB = (long long)Hc * EHc;          // 2,359,296
    const long long QB  = (long long)Ec * Mc * Hc;      // 12,582,912

    f16* p = (f16*)d_ws;
    f16* xb  = p;            p += XB;
    f16* wqb = p;            p += WEB;
    f16* wkb = p;            p += WEB;
    f16* wvb = p;            p += WEB;
    f16* wob = p;            p += WEB;
    f16* wgb = p;            p += WMB;
    f16* wub = p;            p += WMB;
    f16* qb  = p;            p += QB;
    f16* kb  = p;            p += QB;
    f16* vb  = p;            p += QB;
    f16* vtb = p;            p += QB;
    f16* ob    = vb;  // V (un-transposed) dead after transpose -> reuse for attention output
    f16* fused = qb;  // Q dead after attention -> reuse for fused expert outputs

    const int CT = 256;
    // conversions
    cvt_f32_f16_kernel<<<2048, CT, 0, stream>>>(x,      xb,  XB);
    cvt_f32_f16_kernel<<<2048, CT, 0, stream>>>(wq,     wqb, WEB);
    cvt_f32_f16_kernel<<<2048, CT, 0, stream>>>(wk,     wkb, WEB);
    cvt_f32_f16_kernel<<<2048, CT, 0, stream>>>(wv,     wvb, WEB);
    cvt_f32_f16_kernel<<<2048, CT, 0, stream>>>(wo,     wob, WEB);
    cvt_f32_f16_kernel<<<2048, CT, 0, stream>>>(w_gate, wgb, WMB);
    cvt_f32_f16_kernel<<<2048, CT, 0, stream>>>(w_up,   wub, WMB);

    // Q/K/V projections: [4096,768] x [768,768]^T per expert (grid.z = expert)
    dim3 ggrid(Hc / 64, Mc / 32, Ec);
    const long long sW = (long long)Hc * Hc;           // expert stride in weights
    const long long sC = (long long)Mc * Hc;           // expert stride in Q/K/V
    gemm_f16_kernel<<<ggrid, CT, 0, stream>>>(xb, wqb, qb, Hc, Hc, Hc, Hc, 0, sW, sC);
    gemm_f16_kernel<<<ggrid, CT, 0, stream>>>(xb, wkb, kb, Hc, Hc, Hc, Hc, 0, sW, sC);
    gemm_f16_kernel<<<ggrid, CT, 0, stream>>>(xb, wvb, vb, Hc, Hc, Hc, Hc, 0, sW, sC);

    // V transpose for contiguous key-dim access in P*V
    transpose_v_kernel<<<4096, CT, 0, stream>>>(vb, vtb);

    // sliding-window attention: E*B*NH*(S/16) = 12288 waves, 4 waves/block
    attn_kernel<<<(Ec * Bc * NHc * (Sc / 16)) / 4, 128, 0, stream>>>(qb, kb, vtb, ob);

    // per-expert output projection into fused [M, E*H] (column offset via sC = 768)
    gemm_f16_kernel<<<ggrid, CT, 0, stream>>>(ob, wob, fused, Hc, Hc, Hc, EHc, sC, sW, (long long)Hc);

    // gated MLP epilogue -> f32 output
    dim3 mgrid(Hc / 64, Mc / 32, 1);
    mlp_kernel<<<mgrid, CT, 0, stream>>>(fused, wgb, wub, out);
}
